// GCN_61435212202423
// MI455X (gfx1250) — matrix operand order repaired and checked by reading, not compiled, not run on hardware
//
#include <hip/hip_runtime.h>
#include <hip/hip_bf16.h>

typedef __attribute__((ext_vector_type(16))) __bf16 v16bf;
typedef __attribute__((ext_vector_type(8)))  float  v8f;

static __device__ __forceinline__ __bf16 f2bf(float f) {
    return static_cast<__bf16>(f);   // lowers to v_cvt_pk_bf16_f32 on gfx1250
}

// ---------------- utility: zero a dword range ----------------
__global__ void gcn_zero_u32(unsigned* __restrict__ p, long long n) {
    long long t = (long long)blockIdx.x * blockDim.x + threadIdx.x;
    if (t < n) p[t] = 0u;
}

// ---------------- degrees via native u32 atomics ----------------
__global__ void gcn_degree(const int* __restrict__ src, const int* __restrict__ dst,
                           unsigned* __restrict__ dsrc, unsigned* __restrict__ ddst, int E) {
    int t = blockIdx.x * blockDim.x + threadIdx.x;
    if (t < E) {
        atomicAdd(&dsrc[src[t]], 1u);
        atomicAdd(&ddst[dst[t]], 1u);
    }
}

__global__ void gcn_norms(const unsigned* __restrict__ dsrc, const unsigned* __restrict__ ddst,
                          float* __restrict__ onorm, float* __restrict__ inorm, int n) {
    int i = blockIdx.x * blockDim.x + threadIdx.x;
    if (i < n) {
        onorm[i] = rsqrtf(fmaxf((float)dsrc[i], 1.0f));
        inorm[i] = rsqrtf(fmaxf((float)ddst[i], 1.0f));
    }
}

// ---------------- SpMM: m[dst] += h[src] * out_norm[src] ----------------
// One thread per (edge, 4-feature chunk): float4 gather + 4 hw f32 atomics.
// Feature matrix (25.6 MB) is L2-resident on MI455X (192 MB L2), so this
// runs at L2 bandwidth, and f32 atomic-adds resolve in the L2 atomic units.
__global__ void gcn_spmm(const float* __restrict__ h, const float* __restrict__ onorm,
                         const int* __restrict__ src, const int* __restrict__ dst,
                         float* __restrict__ m, int E, int d) {
    const int chunks = d >> 2;
    long long t = (long long)blockIdx.x * blockDim.x + threadIdx.x;
    long long total = (long long)E * chunks;
    if (t >= total) return;
    int e = (int)(t / chunks);
    int c = (int)(t - (long long)e * chunks);
    int s  = src[e];
    int dd = dst[e];
    float w = onorm[s];
    const float4 v = reinterpret_cast<const float4*>(h + (size_t)s * d)[c];
    float* out = m + (size_t)dd * d + (size_t)c * 4;
    unsafeAtomicAdd(out + 0, v.x * w);
    unsafeAtomicAdd(out + 1, v.y * w);
    unsafeAtomicAdd(out + 2, v.z * w);
    unsafeAtomicAdd(out + 3, v.w * w);
}

// ---------------- W pre-pack: f32 row-major -> bf16 in WMMA B-operand order --
// Layout: for column-tile ct (16 cols), k-step kk (32 rows of W), lane L (0..31),
// 16 contiguous bf16 at ((ct*(K/32)+kk)*32 + L)*16 hold
//   B[K = kk*32 + (L>>4)*16 + i][N = ct*16 + (L&15)],  i = 0..15
// so the GEMM loads each B tile as one aligned 32-byte vector per lane.
template <int K>
__global__ void gcn_pack_w(const float* __restrict__ W, __bf16* __restrict__ Wb, int N) {
    constexpr int KT = K / 32;
    int idx = blockIdx.x * blockDim.x + threadIdx.x;   // one bf16 element
    if (idx >= N * K) return;
    const int i    = idx & 15;
    const int lane = (idx >> 4) & 31;
    const int kk   = (idx >> 9) % KT;
    const int ct   = (idx >> 9) / KT;
    const int srcK = kk * 32 + (lane >> 4) * 16 + i;
    const int srcN = ct * 16 + (lane & 15);
    Wb[idx] = f2bf(W[(size_t)srcK * N + srcN]);
}

// ---------------- GEMM: out = relu?( (A * in_norm) @ W + b ) ----------------
// One wave per 16x64 output tile: A tile (16x32 bf16) converted in-register
// once per k-step and reused by NT=4 v_wmma_f32_16x16x32_bf16 ops; B tiles
// come pre-swizzled as single 32-byte vector loads.  f32 accumulate.
template <int K>
__global__ void __launch_bounds__(32)
gcn_gemm_wmma(const float* __restrict__ A, const float* __restrict__ inorm,
              const v16bf* __restrict__ Bp, const float* __restrict__ bias,
              float* __restrict__ out, int n, int N, int relu) {
    constexpr int NT = 4;        // 4 column tiles of 16 -> 16x64 per wave
    constexpr int KT = K / 32;
    const int lane = threadIdx.x;
    const int half = lane >> 4;  // 0: lanes 0-15, 1: lanes 16-31
    const int l16  = lane & 15;

    const int rowA    = blockIdx.x * 16 + l16;   // A row this lane loads
    const int tile0   = blockIdx.y * NT;         // first column tile
    const float nrm = (rowA < n) ? inorm[rowA] : 0.0f;
    const float* Arow = A + (size_t)rowA * K;

    v8f acc[NT] = {};
#pragma unroll
    for (int kk = 0; kk < KT; ++kk) {
        // --- A tile (16x32 bf16, element i holds k = 16*half + i), scaled by in_norm ---
        v16bf a;
        const int ka = kk * 32 + half * 16;
        if (rowA < n) {
#pragma unroll
            for (int i = 0; i < 8; ++i) a[i]     = f2bf(Arow[ka + i] * nrm);
#pragma unroll
            for (int i = 0; i < 8; ++i) a[8 + i] = f2bf(Arow[ka + 8 + i] * nrm);
        } else {
#pragma unroll
            for (int i = 0; i < 16; ++i) a[i] = f2bf(0.0f);
        }
        // --- NT pre-swizzled B tiles: one 32B vector load each ---
#pragma unroll
        for (int t = 0; t < NT; ++t) {
            const v16bf b = Bp[(size_t)((tile0 + t) * KT + kk) * 32 + lane];
            acc[t] = __builtin_amdgcn_wmma_f32_16x16x32_bf16(
                /*neg_a=*/false, a, /*neg_b=*/false, b,
                /*c_mod=*/(short)0, acc[t], /*reuse_a=*/false, /*reuse_b=*/false);
        }
    }

#pragma unroll
    for (int t = 0; t < NT; ++t) {
        const int col = (tile0 + t) * 16 + l16;
        const float bv = bias[col];
#pragma unroll
        for (int r = 0; r < 8; ++r) {
            int row = blockIdx.x * 16 + half * 8 + r;  // C/D layout: M = 8*half + r
            if (row < n) {
                float v = acc[t][r] + bv;
                if (relu) v = fmaxf(v, 0.0f);
                out[(size_t)row * N + col] = v;
            }
        }
    }
}

extern "C" void kernel_launch(void* const* d_in, const int* in_sizes, int n_in,
                              void* d_out, int out_size, void* d_ws, size_t ws_size,
                              hipStream_t stream) {
    const float* x    = (const float*)d_in[0];
    const int*   esrc = (const int*)  d_in[1];
    const int*   edst = (const int*)  d_in[2];
    const float* W0   = (const float*)d_in[3];
    const float* b0   = (const float*)d_in[4];
    const float* W1   = (const float*)d_in[5];
    const float* b1   = (const float*)d_in[6];
    const float* W2   = (const float*)d_in[7];
    const float* b2   = (const float*)d_in[8];

    const int hid  = in_sizes[4];            // 128
    const int inf  = in_sizes[3] / hid;      // 128
    const int ncls = in_sizes[8];            // 64
    const int n    = in_sizes[0] / inf;      // 50000
    const int E    = in_sizes[1];            // 800000
    const int dmax = (inf > hid) ? inf : hid;

    // workspace layout
    unsigned* dsrc  = (unsigned*)d_ws;                   // n u32
    unsigned* ddst  = dsrc + n;                          // n u32
    float*    onorm = (float*)(ddst + n);                // n f32
    float*    inorm = onorm + n;                         // n f32
    float*    mbuf  = inorm + n;                         // n*dmax f32
    float*    hbuf  = mbuf + (size_t)n * dmax;           // n*dmax f32
    __bf16*   wbuf  = (__bf16*)(hbuf + (size_t)n * dmax); // dmax*dmax bf16 (32B-aligned)

    const int ZT = 256;
    auto zero = [&](unsigned* p, long long cnt) {
        long long blocks = (cnt + ZT - 1) / ZT;
        gcn_zero_u32<<<(unsigned)blocks, ZT, 0, stream>>>(p, cnt);
    };

    // ---- degrees + norms ----
    zero(dsrc, 2LL * n);
    gcn_degree<<<(E + 255) / 256, 256, 0, stream>>>(esrc, edst, dsrc, ddst, E);
    gcn_norms<<<(n + 255) / 256, 256, 0, stream>>>(dsrc, ddst, onorm, inorm, n);

    auto layer = [&](const float* hin, int din, const float* W, const float* b,
                     int dout, float* hout, int relu) {
        zero((unsigned*)mbuf, (long long)n * din);
        long long sp_total = (long long)E * (din >> 2);
        gcn_spmm<<<(unsigned)((sp_total + 255) / 256), 256, 0, stream>>>(
            hin, onorm, esrc, edst, mbuf, E, din);
        gcn_pack_w<128><<<(dout * din + 255) / 256, 256, 0, stream>>>(W, wbuf, dout);
        dim3 grid((n + 15) / 16, dout / 64);   // 16x64 tile per wave
        gcn_gemm_wmma<128><<<grid, 32, 0, stream>>>(
            mbuf, inorm, (const v16bf*)wbuf, b, hout, n, dout, relu);
    };

    layer(x,    inf, W0, b0, hid,  hbuf,           1);
    layer(hbuf, hid, W1, b1, hid,  hbuf,           1);
    layer(hbuf, hid, W2, b2, ncls, (float*)d_out,  0);
}
